// AfmoeAttention_54898271977520
// MI455X (gfx1250) — matrix operand order repaired and checked by reading, not compiled, not run on hardware
//
#include <hip/hip_runtime.h>
#include <hip/hip_bf16.h>

#define DEV __device__ __forceinline__

typedef __attribute__((ext_vector_type(16))) __bf16 v16bf;
typedef __attribute__((ext_vector_type(8)))  __bf16 v8bf;
typedef __attribute__((ext_vector_type(8)))  float  v8f;

constexpr int Bb   = 2;
constexpr int Ss   = 2048;
constexpr int HIDc = 2048;
constexpr int Hh   = 16;
constexpr int HKk  = 4;
constexpr int Dd   = 128;
constexpr int WIN  = 1024;
constexpr int MTOK = Bb * Ss;   // 4096 token rows

// ---------------------------------------------------------------------------
// WMMA fragment helpers (CDNA5 ISA 7.12.2 layouts, wave32)
// A fragment 16x32 bf16: lane M = lane%16, hi = lane/16.
//   elements = A[M][k0+8*hi .. +7] ++ A[M][k0+16+8*hi .. +7]
// B fragment 32x16 bf16, operand stored as N x K row-major:
//   elements = W[N][k0+8*hi .. +7] ++ W[N][k0+16+8*hi .. +7]   (N = lane%16)
// C/D 16x16 f32: lane holds col N=lane%16, reg r holds row M = r + 8*hi.
// ---------------------------------------------------------------------------
DEV v16bf load_a_frag(const __bf16* base, int row, int ldk, int k0, int hi) {
  const __bf16* p = base + (long)row * ldk + k0 + 8 * hi;
  v8bf lo = *(const v8bf*)p;
  v8bf hv = *(const v8bf*)(p + 16);
  v16bf r;
#pragma unroll
  for (int i = 0; i < 8; ++i) { r[i] = lo[i]; r[i + 8] = hv[i]; }
  return r;
}

DEV v16bf load_b_frag(const __bf16* base, int ncol, int ldk, int k0, int hi) {
  const __bf16* p = base + (long)ncol * ldk + k0 + 8 * hi;
  v8bf lo = *(const v8bf*)p;
  v8bf hv = *(const v8bf*)(p + 16);
  v16bf r;
#pragma unroll
  for (int i = 0; i < 8; ++i) { r[i] = lo[i]; r[i + 8] = hv[i]; }
  return r;
}

DEV v8f wmma_bf16(v16bf a, v16bf b, v8f c) {
  return __builtin_amdgcn_wmma_f32_16x16x32_bf16(
      /*neg_a=*/false, a, /*neg_b=*/false, b,
      /*c_mod=*/(short)0, c, /*reuse_a=*/false, /*reuse_b=*/false);
}

// ---------------------------------------------------------------------------
// Async global -> LDS copy (CDNA5 GLOBAL_LOAD_ASYNC_TO_LDS_B128, ASYNCcnt)
// ---------------------------------------------------------------------------
DEV void async_copy_b128(unsigned lds_off, const __bf16* g) {
  asm volatile("global_load_async_to_lds_b128 %0, %1, off"
               :: "v"(lds_off), "v"(g) : "memory");
}
DEV void wait_async_le3() {
#if __has_builtin(__builtin_amdgcn_s_wait_asynccnt)
  __builtin_amdgcn_s_wait_asynccnt(3);
#else
  asm volatile("s_wait_asynccnt 0x3" ::: "memory");
#endif
}
DEV void wait_async_le0() {
#if __has_builtin(__builtin_amdgcn_s_wait_asynccnt)
  __builtin_amdgcn_s_wait_asynccnt(0);
#else
  asm volatile("s_wait_asynccnt 0x0" ::: "memory");
#endif
}

// ---------------------------------------------------------------------------
// f32 -> bf16 convert (grid-stride)
// ---------------------------------------------------------------------------
__global__ __launch_bounds__(256) void cvt_f32_bf16(const float* __restrict__ in,
                                                    __bf16* __restrict__ out, long n) {
  long i = (long)blockIdx.x * blockDim.x + threadIdx.x;
  long stride = (long)gridDim.x * blockDim.x;
  for (; i < n; i += stride) out[i] = (__bf16)in[i];
}

// ---------------------------------------------------------------------------
// C[M,N](f32) = A[M,K](bf16) @ W[N,K](bf16)^T
// 256 threads = 8 waves; block tile 64(M) x 128(N); each wave does 32x32.
// Double-buffered async global->LDS staging:
//   As: 2 x (64 x 32)  bf16 (4 KB each),  Bs: 2 x (128 x 32) bf16 (8 KB each)
// Each thread issues exactly 3 b128 async copies per k-tile (1 A + 2 B), so
// "s_wait_asynccnt <= 3" retires the previous tile while the prefetch flies.
// ---------------------------------------------------------------------------
__global__ __launch_bounds__(256) void gemm_bf16nt(const __bf16* __restrict__ A,
                                                   const __bf16* __restrict__ W,
                                                   float* __restrict__ C,
                                                   int M, int N, int K) {
  __shared__ __bf16 As[2][64 * 32];
  __shared__ __bf16 Bs[2][128 * 32];
  int t = threadIdx.x;
  int lane = t & 31, wave = t >> 5;
  int hi = lane >> 4, l16 = lane & 15;
  int mb = blockIdx.y * 64;
  int nb = blockIdx.x * 128;
  int mw = (wave >> 2) * 32;
  int nw = (wave & 3) * 32;

  int cr = t >> 2;             // copy row 0..63
  int ck = (t & 3) * 8;        // copy k offset 0/8/16/24

  // LDS aperture base has zero low 32 bits, so the truncated flat address is
  // the LDS byte offset the async instruction's VDST operand expects.
  unsigned As_off[2] = { (unsigned)(uintptr_t)&As[0][0], (unsigned)(uintptr_t)&As[1][0] };
  unsigned Bs_off[2] = { (unsigned)(uintptr_t)&Bs[0][0], (unsigned)(uintptr_t)&Bs[1][0] };

  auto issue = [&](int buf, int k0) {
    async_copy_b128(As_off[buf] + (unsigned)(cr * 32 + ck) * 2,
                    A + (long)(mb + cr) * K + k0 + ck);
    async_copy_b128(Bs_off[buf] + (unsigned)(cr * 32 + ck) * 2,
                    W + (long)(nb + cr) * K + k0 + ck);
    async_copy_b128(Bs_off[buf] + (unsigned)((cr + 64) * 32 + ck) * 2,
                    W + (long)(nb + 64 + cr) * K + k0 + ck);
  };

  v8f acc00 = {}, acc01 = {}, acc10 = {}, acc11 = {};
  int nit = K / 32;
  issue(0, 0);
  for (int it = 0; it < nit; ++it) {
    int buf = it & 1;
    if (it + 1 < nit) { issue(buf ^ 1, (it + 1) * 32); wait_async_le3(); }
    else              { wait_async_le0(); }
    __syncthreads();
    const __bf16* Ab = &As[buf][0];
    const __bf16* Bbp = &Bs[buf][0];
    v16bf a0 = load_a_frag(Ab,  mw + l16,      32, 0, hi);
    v16bf a1 = load_a_frag(Ab,  mw + 16 + l16, 32, 0, hi);
    v16bf b0 = load_b_frag(Bbp, nw + l16,      32, 0, hi);
    v16bf b1 = load_b_frag(Bbp, nw + 16 + l16, 32, 0, hi);
    acc00 = wmma_bf16(a0, b0, acc00);
    acc01 = wmma_bf16(a0, b1, acc01);
    acc10 = wmma_bf16(a1, b0, acc10);
    acc11 = wmma_bf16(a1, b1, acc11);
    __syncthreads();
  }
  int m0 = mb + mw, n0 = nb + nw;
#pragma unroll
  for (int r = 0; r < 8; ++r) {
    int mr0 = m0 + r + 8 * hi;
    int mr1 = mr0 + 16;
    C[(long)mr0 * N + n0 + l16]      = acc00[r];
    C[(long)mr0 * N + n0 + 16 + l16] = acc01[r];
    C[(long)mr1 * N + n0 + l16]      = acc10[r];
    C[(long)mr1 * N + n0 + 16 + l16] = acc11[r];
  }
}

// ---------------------------------------------------------------------------
// Per-head RMSNorm + RoPE. src: [B*S, NH*D] f32.  dst: [B, NH, S, D] bf16.
// One 64-thread block per (b, s, h); thread t owns the rotary pair (2t, 2t+1).
// ---------------------------------------------------------------------------
__global__ __launch_bounds__(64) void rmsnorm_rope(const float* __restrict__ src,
                                                   const float* __restrict__ fc,
                                                   const float* __restrict__ w,
                                                   __bf16* __restrict__ dst, int NH) {
  __shared__ float red[64];
  int t = threadIdx.x;
  int h = blockIdx.x % NH;
  int s = (blockIdx.x / NH) % Ss;
  int b = blockIdx.x / (NH * Ss);
  const float* p = src + ((long)(b * Ss + s) * NH + h) * Dd;
  float x1 = p[2 * t], x2 = p[2 * t + 1];
  red[t] = x1 * x1 + x2 * x2;
  __syncthreads();
#pragma unroll
  for (int off = 32; off > 0; off >>= 1) {
    if (t < off) red[t] += red[t + off];
    __syncthreads();
  }
  float inv = rsqrtf(red[0] * (1.0f / Dd) + 1e-6f);
  float c  = fc[((long)s * (Dd / 2) + t) * 2 + 0];
  float sn = fc[((long)s * (Dd / 2) + t) * 2 + 1];
  float y1 = x1 * inv * w[2 * t];
  float y2 = x2 * inv * w[2 * t + 1];
  __bf16* q = dst + ((long)(b * NH + h) * Ss + s) * Dd;
  q[2 * t]     = (__bf16)(y1 * c - y2 * sn);
  q[2 * t + 1] = (__bf16)(y1 * sn + y2 * c);
}

// ---------------------------------------------------------------------------
// V: [B*S, HK*D] f32  ->  Vt: [B, HK, D, S] bf16 (transposed so P@V B-frags
// are contiguous b128 loads along the key axis)
// ---------------------------------------------------------------------------
__global__ __launch_bounds__(256) void v_transpose(const float* __restrict__ vf,
                                                   __bf16* __restrict__ vt) {
  long i = (long)blockIdx.x * blockDim.x + threadIdx.x;
  long total = (long)Bb * Ss * HKk * Dd;
  if (i >= total) return;
  int d  = (int)(i % Dd);
  int kh = (int)((i / Dd) % HKk);
  int s  = (int)((i / ((long)Dd * HKk)) % Ss);
  int b  = (int)(i / ((long)Dd * HKk * Ss));
  vt[((long)(b * HKk + kh) * Dd + d) * Ss + s] = (__bf16)vf[i];
}

// ---------------------------------------------------------------------------
// Flash-style sliding-window GQA attention + sigmoid gating.
// One wave per (b, h, 16-query tile). Online softmax over 32-key chunks.
// Qb: [B,H,S,D] bf16   Kb: [B,HK,S,D] bf16   Vt: [B,HK,D,S] bf16
// gate: [B*S, H*D] f32 -> Ob: [B*S, H*D] bf16
// ---------------------------------------------------------------------------
__global__ __launch_bounds__(32) void attn_kernel(const __bf16* __restrict__ Qb,
                                                  const __bf16* __restrict__ Kb,
                                                  const __bf16* __restrict__ Vt,
                                                  const float* __restrict__ gate,
                                                  __bf16* __restrict__ Ob) {
  __shared__ __bf16 Psh[16 * 32];
  int lane = threadIdx.x;
  int hi = lane >> 4, l16 = lane & 15;
  int q0 = blockIdx.x * 16;
  int h  = blockIdx.y;
  int b  = blockIdx.z;
  int kvh = h / (Hh / HKk);
  const __bf16* Q  = Qb + ((long)(b * Hh  + h)   * Ss) * Dd;
  const __bf16* Km = Kb + ((long)(b * HKk + kvh) * Ss) * Dd;
  const __bf16* Vm = Vt + ((long)(b * HKk + kvh) * Dd) * Ss;

  v16bf qf[4];
#pragma unroll
  for (int i = 0; i < 4; ++i) qf[i] = load_a_frag(Q, q0 + l16, Dd, 32 * i, hi);

  v8f acc[8] = {};
  float mI[8], lI[8];
#pragma unroll
  for (int r = 0; r < 8; ++r) { mI[r] = -3.0e38f; lI[r] = 0.0f; }

  const float scale = 0.088388347648318447f;  // 1/sqrt(128)
  int kstart = (q0 > WIN) ? ((q0 - WIN) & ~31) : 0;

  for (int kc = kstart; kc < q0 + 16; kc += 32) {
    v8f s0 = {}, s1 = {};
#pragma unroll
    for (int dk = 0; dk < 4; ++dk) {
      v16bf b0 = load_b_frag(Km, kc + l16,      Dd, 32 * dk, hi);
      v16bf b1 = load_b_frag(Km, kc + 16 + l16, Dd, 32 * dk, hi);
      s0 = wmma_bf16(qf[dk], b0, s0);
      s1 = wmma_bf16(qf[dk], b1, s1);
    }
#pragma unroll
    for (int r = 0; r < 8; ++r) {
      int qi  = q0 + r + 8 * hi;
      int k0i = kc + l16;
      int k1i = k0i + 16;
      bool ok0 = (k0i <= qi) && (qi - k0i <= WIN);
      bool ok1 = (k1i <= qi) && (qi - k1i <= WIN);
      float v0 = ok0 ? s0[r] * scale : -3.0e38f;
      float v1 = ok1 ? s1[r] * scale : -3.0e38f;
      float mx = fmaxf(v0, v1);
#pragma unroll
      for (int off = 1; off < 16; off <<= 1) mx = fmaxf(mx, __shfl_xor(mx, off));
      float mnew  = fmaxf(mI[r], mx);
      float alpha = __expf(mI[r] - mnew);
      float p0 = ok0 ? __expf(v0 - mnew) : 0.0f;
      float p1 = ok1 ? __expf(v1 - mnew) : 0.0f;
      float ps = p0 + p1;
#pragma unroll
      for (int off = 1; off < 16; off <<= 1) ps += __shfl_xor(ps, off);
      lI[r] = lI[r] * alpha + ps;
      mI[r] = mnew;
#pragma unroll
      for (int n = 0; n < 8; ++n) acc[n][r] *= alpha;
      Psh[(r + 8 * hi) * 32 + l16]      = (__bf16)p0;
      Psh[(r + 8 * hi) * 32 + 16 + l16] = (__bf16)p1;
    }
    __syncthreads();
    v16bf pf = load_a_frag(Psh, l16, 32, 0, hi);
#pragma unroll
    for (int n = 0; n < 8; ++n) {
      v16bf bv = load_b_frag(Vm, n * 16 + l16, Ss, kc, hi);
      acc[n] = wmma_bf16(pf, bv, acc[n]);
    }
    __syncthreads();
  }

#pragma unroll
  for (int n = 0; n < 8; ++n) {
#pragma unroll
    for (int r = 0; r < 8; ++r) {
      int qi = q0 + r + 8 * hi;
      int dc = n * 16 + l16;
      long gi = (long)(b * Ss + qi) * (Hh * Dd) + h * Dd + dc;
      float o = acc[n][r] / lI[r];
      float g = gate[gi];
      o *= 1.0f / (1.0f + __expf(-g));
      Ob[gi] = (__bf16)o;
    }
  }
}

// ---------------------------------------------------------------------------
// Host-side orchestration
// ---------------------------------------------------------------------------
extern "C" void kernel_launch(void* const* d_in, const int* in_sizes, int n_in,
                              void* d_out, int out_size, void* d_ws, size_t ws_size,
                              hipStream_t stream) {
  (void)in_sizes; (void)n_in; (void)out_size; (void)ws_size;
  const float* x   = (const float*)d_in[0];
  const float* fc  = (const float*)d_in[1];
  const float* wq  = (const float*)d_in[2];
  const float* wk  = (const float*)d_in[3];
  const float* wv  = (const float*)d_in[4];
  const float* wo  = (const float*)d_in[5];
  const float* wg  = (const float*)d_in[6];
  const float* qnw = (const float*)d_in[7];
  const float* knw = (const float*)d_in[8];
  float* out = (float*)d_out;

  char* wsp = (char*)d_ws;
  auto alloc = [&](size_t bytes) -> char* {
    char* p = wsp;
    wsp += (bytes + 255) & ~(size_t)255;
    return p;
  };
  __bf16* xb  = (__bf16*)alloc((size_t)MTOK * HIDc * 2);       // x bf16
  __bf16* wqb = (__bf16*)alloc((size_t)Hh * Dd * HIDc * 2);
  __bf16* wkb = (__bf16*)alloc((size_t)HKk * Dd * HIDc * 2);
  __bf16* wvb = (__bf16*)alloc((size_t)HKk * Dd * HIDc * 2);
  __bf16* wgb = (__bf16*)alloc((size_t)Hh * Dd * HIDc * 2);
  __bf16* wob = (__bf16*)alloc((size_t)HIDc * Hh * Dd * 2);
  float*  qf  = (float*)alloc((size_t)MTOK * Hh * Dd * 4);     // q pre-norm f32
  float*  kf  = (float*)alloc((size_t)MTOK * HKk * Dd * 4);
  float*  vf  = (float*)alloc((size_t)MTOK * HKk * Dd * 4);
  float*  gf  = (float*)alloc((size_t)MTOK * Hh * Dd * 4);     // gate f32
  __bf16* qbb = (__bf16*)alloc((size_t)Bb * Hh * Ss * Dd * 2); // [B,H,S,D]
  __bf16* kbb = (__bf16*)alloc((size_t)Bb * HKk * Ss * Dd * 2);// [B,HK,S,D]
  __bf16* vtb = (__bf16*)alloc((size_t)Bb * HKk * Dd * Ss * 2);// [B,HK,D,S]
  __bf16* ob  = (__bf16*)alloc((size_t)MTOK * Hh * Dd * 2);    // gated attn bf16

  auto cvt = [&](const float* in, __bf16* o, long n) {
    long nb = (n + 255) / 256;
    int blocks = nb > 16384 ? 16384 : (int)nb;
    cvt_f32_bf16<<<dim3(blocks), dim3(256), 0, stream>>>(in, o, n);
  };
  cvt(x,  xb,  (long)MTOK * HIDc);
  cvt(wq, wqb, (long)Hh * Dd * HIDc);
  cvt(wk, wkb, (long)HKk * Dd * HIDc);
  cvt(wv, wvb, (long)HKk * Dd * HIDc);
  cvt(wg, wgb, (long)Hh * Dd * HIDc);
  cvt(wo, wob, (long)HIDc * Hh * Dd);

  auto gemm = [&](const __bf16* A, const __bf16* W, float* C, int M, int N, int K) {
    gemm_bf16nt<<<dim3(N / 128, M / 64), dim3(256), 0, stream>>>(A, W, C, M, N, K);
  };
  gemm(xb, wqb, qf, MTOK, Hh * Dd,  HIDc);   // q
  gemm(xb, wkb, kf, MTOK, HKk * Dd, HIDc);   // k
  gemm(xb, wvb, vf, MTOK, HKk * Dd, HIDc);   // v
  gemm(xb, wgb, gf, MTOK, Hh * Dd,  HIDc);   // gate

  rmsnorm_rope<<<dim3(Bb * Ss * Hh),  dim3(64), 0, stream>>>(qf, fc, qnw, qbb, Hh);
  rmsnorm_rope<<<dim3(Bb * Ss * HKk), dim3(64), 0, stream>>>(kf, fc, knw, kbb, HKk);
  v_transpose<<<dim3((Bb * Ss * HKk * Dd) / 256), dim3(256), 0, stream>>>(vf, vtb);

  attn_kernel<<<dim3(Ss / 16, Hh, Bb), dim3(32), 0, stream>>>(qbb, kbb, vtb, gf, ob);

  gemm(ob, wob, out, MTOK, HIDc, Hh * Dd);   // final projection
}
